// MyProcessorBlock_71906342470116
// MI455X (gfx1250) — compile-verified, matrix operand order checked by reading, not run on hardware
//
#include <hip/hip_runtime.h>
#include <hip/hip_bf16.h>

// ---------------------------------------------------------------------------
// GNN message-passing block for MI455X (gfx1250, wave32, WMMA bf16).
//
// Three fused 3-layer MLPs (256->128->128->128 + LayerNorm) over edge/node
// rows, computed with v_wmma_f32_16x16x32_bf16 (f32 accumulate). Weights are
// held as bf16 WMMA fragments in VGPRs per wave (one wave per 16-col slice),
// amortized over a grid-stride loop of 16-row tiles. Graph gathers/scatters
// are done in f32 via global_atomic_add_f32.
// ---------------------------------------------------------------------------

#define DEV __device__ __forceinline__

typedef __bf16        bf16x16 __attribute__((ext_vector_type(16)));
typedef float         f32x8   __attribute__((ext_vector_type(8)));
typedef unsigned int  u32x4   __attribute__((ext_vector_type(4)));
typedef unsigned int  u32x2   __attribute__((ext_vector_type(2)));

static constexpr int N_NODES = 50000;
static constexpr int N_EDGES = 600000;
static constexpr int D       = 128;
static constexpr float LN_EPS = 1e-5f;

// Build 16x32 bf16 A-fragment from a row-major bf16 LDS tile (stride S elems).
// ISA layout: lane(0..15)=row M=lane, K = kb+0..7 (v0..3) and kb+16..23 (v4..7)
// with kb = 8*(lane>=16). Each half is 8 contiguous bf16 = one 16B LDS load.
DEV bf16x16 fragA_lds(const __bf16* t, int lane, int k0, int S) {
  int r  = lane & 15;
  int kb = (lane >> 4) * 8;
  const __bf16* p = t + r * S + k0 + kb;
  union { bf16x16 v; u32x4 q[2]; } u;
  u.q[0] = *(const u32x4*)(p);
  u.q[1] = *(const u32x4*)(p + 16);
  return u.v;
}

// Build 32x16 bf16 B-fragment from a row-major f32 weight matrix W[K][128].
// Mirrors the A layout with N in place of M (lane = column).
DEV bf16x16 fragB_w(const float* __restrict__ W, int lane, int k0, int n0) {
  int n  = n0 + (lane & 15);
  int kb = (lane >> 4) * 8;
  bf16x16 v;
#pragma unroll
  for (int j = 0; j < 16; ++j) {
    int k = k0 + kb + j + ((j >= 8) ? 8 : 0);
    v[j] = (__bf16)W[k * D + n];
  }
  return v;
}

DEV f32x8 wmma_bf16(bf16x16 a, bf16x16 b, f32x8 c) {
  // (neg_a, A, neg_b, B, c_mod, C, reuse_a, reuse_b)
  return __builtin_amdgcn_wmma_f32_16x16x32_bf16(false, a, false, b,
                                                 (short)0, c, false, false);
}

// Convert a float4 to 4 packed bf16 and store with a single 8-byte DS write.
DEV void store_bf4(__bf16* d, float4 v) {
  union { __bf16 h[4]; u32x2 u; } p;
  p.h[0] = (__bf16)v.x; p.h[1] = (__bf16)v.y;
  p.h[2] = (__bf16)v.z; p.h[3] = (__bf16)v.w;
  *(u32x2*)d = p.u;      // 8B-aligned by construction
}

// Store a 16x16 f32 accumulator tile (+bias, ReLU) as bf16 into LDS[16][128].
// C/D layout: lane l holds N = l&15, rows M = r + 8*(l>>4) for VGPR r.
DEV void store_hidden(__bf16* H, f32x8 c, int lane, int n0, float bias) {
  int col = n0 + (lane & 15);
  int rb  = (lane >> 4) * 8;
#pragma unroll
  for (int r = 0; r < 8; ++r) {
    float v = fmaxf(c[r] + bias, 0.f);   // v_max_num_f32 ReLU
    H[(rb + r) * D + col] = (__bf16)v;
  }
}

DEV void store_x(float* X, f32x8 c, int lane, int n0, float bias) {
  int col = n0 + (lane & 15);
  int rb  = (lane >> 4) * 8;
#pragma unroll
  for (int r = 0; r < 8; ++r)
    X[(rb + r) * D + col] = c[r] + bias;
}

DEV void atomic_add_f32(float* p, float v) {
  __hip_atomic_fetch_add(p, v, __ATOMIC_RELAXED, __HIP_MEMORY_SCOPE_AGENT);
}

// ---------------------------------------------------------------------------
// MODE 0 (messages): A = [h_nodes[snd[e]], h_edges[e]]
//   epilogue: atomic-add LN output into aggm[rcv[e]];
//   load phase also atomic-adds h_edges[e] into agge[rcv[e]] (segment_sum).
// MODE 1 (nodes):    A = [h_nodes[r], aggm[r]];        out = h_nodes + mlp
// MODE 2 (edges):    A = [h_edges[e], agge[snd[e]] - h_edges[e]]; out = h_edges + mlp
// ---------------------------------------------------------------------------
template <int MODE>
__global__ __launch_bounds__(256)
void gnn_mlp_fused(const float* __restrict__ srcL,
                   const float* __restrict__ srcR,
                   const int*   __restrict__ senders,
                   const int*   __restrict__ receivers,
                   const float* __restrict__ w1, const float* __restrict__ b1,
                   const float* __restrict__ w2, const float* __restrict__ b2,
                   const float* __restrict__ w3, const float* __restrict__ b3,
                   const float* __restrict__ gma, const float* __restrict__ bta,
                   const float* __restrict__ resid,
                   float* __restrict__ out,
                   float* __restrict__ aggm,
                   float* __restrict__ agge,
                   int nTiles) {
  __shared__ __align__(16) __bf16 At[16 * 256];   // layer-1 input
  __shared__ __align__(16) __bf16 H1[16 * D];     // hidden 1
  __shared__ __align__(16) __bf16 H2[16 * D];     // hidden 2
  __shared__ __align__(16) float  X [16 * D];     // pre-LN (f32)
  __shared__ int sSnd[16];
  __shared__ int sRcv[16];

  const int tid  = threadIdx.x;
  const int lane = tid & 31;
  const int wave = tid >> 5;
  const int n0   = wave * 16;   // this wave's 16-column output slice

  // --- Resident weight fragments (128 VGPRs of bf16) ---------------------
  bf16x16 wb1[8], wb2[4], wb3[4];
#pragma unroll
  for (int kk = 0; kk < 8; ++kk) wb1[kk] = fragB_w(w1, lane, kk * 32, n0);
#pragma unroll
  for (int kk = 0; kk < 4; ++kk) wb2[kk] = fragB_w(w2, lane, kk * 32, n0);
#pragma unroll
  for (int kk = 0; kk < 4; ++kk) wb3[kk] = fragB_w(w3, lane, kk * 32, n0);
  const float bias1 = b1[n0 + (lane & 15)];
  const float bias2 = b2[n0 + (lane & 15)];
  const float bias3 = b3[n0 + (lane & 15)];

  // --- loop-invariant LayerNorm params for this lane's 4 columns ----------
  float g4[4], bt4[4];
#pragma unroll
  for (int j = 0; j < 4; ++j) {
    g4[j]  = gma[lane + 32 * j];
    bt4[j] = bta[lane + 32 * j];
  }

  for (int tile = blockIdx.x; tile < nTiles; tile += gridDim.x) {
    const int r0 = tile * 16;

    // stage per-tile graph indices
    if (tid < 16) {
      if (MODE == 0) { sSnd[tid] = senders[r0 + tid]; sRcv[tid] = receivers[r0 + tid]; }
      if (MODE == 2) { sSnd[tid] = senders[r0 + tid]; }
    }
    __syncthreads();

    // speculative prefetch of the next tile's rows (global_prefetch_b8)
    {
      int nt = tile + gridDim.x;
      if (nt < nTiles && MODE != 0) {
        __builtin_prefetch(srcL + (size_t)nt * 16 * D, 0, 1);
        __builtin_prefetch(srcR + (size_t)nt * 16 * D, 0, 1);
      }
    }

    // --- gather the 16x256 f32 input tile -> bf16 LDS ---------------------
    // 16 rows x 64 float4 segments = 1024; 256 threads x 4 iterations.
#pragma unroll
    for (int it = 0; it < 4; ++it) {
      int q   = tid + it * 256;
      int r   = q >> 6;         // tile row 0..15
      int seg = q & 63;         // float4 segment 0..63 (cols seg*4..seg*4+3)
      int row = r0 + r;
      float4 v;
      if (seg < 32) {           // left half: 128 cols
        int li = (MODE == 0) ? sSnd[r] : row;
        v = *(const float4*)(srcL + (size_t)li * D + seg * 4);
      } else {                  // right half: 128 cols
        int c4 = (seg - 32) * 4;
        if (MODE == 0) {
          v = *(const float4*)(srcR + (size_t)row * D + c4);
          // fused segment_sum(h_edges, receivers) -> agge
          float* dst = agge + (size_t)sRcv[r] * D + c4;
          atomic_add_f32(dst + 0, v.x);
          atomic_add_f32(dst + 1, v.y);
          atomic_add_f32(dst + 2, v.z);
          atomic_add_f32(dst + 3, v.w);
        } else if (MODE == 1) {
          v = *(const float4*)(srcR + (size_t)row * D + c4);
        } else {
          float4 a = *(const float4*)(srcR + (size_t)sSnd[r] * D + c4);
          float4 h = *(const float4*)(srcL + (size_t)row * D + c4);
          v.x = a.x - h.x; v.y = a.y - h.y; v.z = a.z - h.z; v.w = a.w - h.w;
        }
      }
      store_bf4(&At[r * 256 + seg * 4], v);
    }
    __syncthreads();

    // --- layer 1: [16x256] @ [256x128] -> relu ----------------------------
    f32x8 c = {0.f, 0.f, 0.f, 0.f, 0.f, 0.f, 0.f, 0.f};
#pragma unroll
    for (int kk = 0; kk < 8; ++kk)
      c = wmma_bf16(fragA_lds(At, lane, kk * 32, 256), wb1[kk], c);
    store_hidden(H1, c, lane, n0, bias1);
    __syncthreads();

    // --- layer 2: [16x128] @ [128x128] -> relu ----------------------------
    c = f32x8{0.f, 0.f, 0.f, 0.f, 0.f, 0.f, 0.f, 0.f};
#pragma unroll
    for (int kk = 0; kk < 4; ++kk)
      c = wmma_bf16(fragA_lds(H1, lane, kk * 32, D), wb2[kk], c);
    store_hidden(H2, c, lane, n0, bias2);
    __syncthreads();

    // --- layer 3: [16x128] @ [128x128] + bias (pre-LN, f32) ---------------
    c = f32x8{0.f, 0.f, 0.f, 0.f, 0.f, 0.f, 0.f, 0.f};
#pragma unroll
    for (int kk = 0; kk < 4; ++kk)
      c = wmma_bf16(fragA_lds(H2, lane, kk * 32, D), wb3[kk], c);
    store_x(X, c, lane, n0, bias3);
    __syncthreads();

    // --- LayerNorm + epilogue: wave w owns rows 2w, 2w+1 ------------------
#pragma unroll
    for (int rr = 0; rr < 2; ++rr) {
      int row = wave * 2 + rr;
      float x[4], s = 0.f, ssq = 0.f;
#pragma unroll
      for (int j = 0; j < 4; ++j) {
        x[j] = X[row * D + lane + 32 * j];
        s += x[j]; ssq += x[j] * x[j];
      }
#pragma unroll
      for (int m = 16; m >= 1; m >>= 1) {   // wave32 reduction
        s   += __shfl_xor(s, m, 32);
        ssq += __shfl_xor(ssq, m, 32);
      }
      float mu  = s * (1.f / 128.f);
      float var = ssq * (1.f / 128.f) - mu * mu;
      float rs  = rsqrtf(var + LN_EPS);
#pragma unroll
      for (int j = 0; j < 4; ++j) {
        int col = lane + 32 * j;
        float y = (x[j] - mu) * rs * g4[j] + bt4[j];
        if (MODE == 0) {
          atomic_add_f32(&aggm[(size_t)sRcv[row] * D + col], y);
        } else {
          size_t o = (size_t)(r0 + row) * D + col;
          out[o] = resid[o] + y;
        }
      }
    }
    __syncthreads();   // protect LDS reuse across tiles
  }
}

__global__ void gnn_zero_ws(float4* __restrict__ p, int n4) {
  int i = blockIdx.x * blockDim.x + threadIdx.x;
  if (i < n4) p[i] = make_float4(0.f, 0.f, 0.f, 0.f);
}

// ---------------------------------------------------------------------------
// Host launcher. Inputs (pytree-flattened, setup_inputs order):
//   [0] h_nodes [N,128] f32       [1] h_edges [E,128] f32
//   [2] edge_index [2,E] int32
//   [3..10]  msg_params  (w1,b1,w2,b2,w3,b3,gamma,beta)
//   [11..18] node_params         [19..26] edge_params
// d_out = h_nodes_new (N*128) ++ h_edges_new (E*128), f32.
// d_ws  = aggm (N*128 f32) ++ agge (N*128 f32).
// ---------------------------------------------------------------------------
extern "C" void kernel_launch(void* const* d_in, const int* in_sizes, int n_in,
                              void* d_out, int out_size, void* d_ws, size_t ws_size,
                              hipStream_t stream) {
  const float* h_nodes = (const float*)d_in[0];
  const float* h_edges = (const float*)d_in[1];
  const int*   eidx    = (const int*)d_in[2];
  const float* mp[8]; const float* np[8]; const float* ep[8];
  for (int i = 0; i < 8; ++i) {
    mp[i] = (const float*)d_in[3 + i];
    np[i] = (const float*)d_in[11 + i];
    ep[i] = (const float*)d_in[19 + i];
  }
  const int* senders   = eidx;
  const int* receivers = eidx + N_EDGES;

  float* aggm = (float*)d_ws;                       // [N,128]
  float* agge = aggm + (size_t)N_NODES * D;         // [N,128]
  float* outN = (float*)d_out;                      // [N,128]
  float* outE = outN + (size_t)N_NODES * D;         // [E,128]

  // zero both aggregation buffers
  {
    int n4 = (2 * N_NODES * D) / 4;
    gnn_zero_ws<<<(n4 + 255) / 256, 256, 0, stream>>>((float4*)d_ws, n4);
  }

  const int tilesE = N_EDGES / 16;   // 37500
  const int tilesN = N_NODES / 16;   // 3125
  const int gridE  = tilesE < 4096 ? tilesE : 4096;
  const int gridN  = tilesN < 4096 ? tilesN : 4096;

  // 1) messages = MLP([h_nodes[snd], h_edges]); scatter-add into aggm;
  //    fused scatter-add of h_edges into agge.
  gnn_mlp_fused<0><<<gridE, 256, 0, stream>>>(
      h_nodes, h_edges, senders, receivers,
      mp[0], mp[1], mp[2], mp[3], mp[4], mp[5], mp[6], mp[7],
      nullptr, nullptr, aggm, agge, tilesE);

  // 2) h_nodes_new = h_nodes + MLP([h_nodes, aggm])
  gnn_mlp_fused<1><<<gridN, 256, 0, stream>>>(
      h_nodes, aggm, nullptr, nullptr,
      np[0], np[1], np[2], np[3], np[4], np[5], np[6], np[7],
      h_nodes, outN, nullptr, nullptr, tilesN);

  // 3) h_edges_new = h_edges + MLP([h_edges, agge[snd] - h_edges])
  gnn_mlp_fused<2><<<gridE, 256, 0, stream>>>(
      h_edges, agge, senders, nullptr,
      ep[0], ep[1], ep[2], ep[3], ep[4], ep[5], ep[6], ep[7],
      h_edges, outE, nullptr, nullptr, tilesE);
}
